// NgramMultiheadAttention_7894149890118
// MI455X (gfx1250) — matrix operand order repaired and checked by reading, not compiled, not run on hardware
//
#include <hip/hip_runtime.h>
#include <hip/hip_bf16.h>
#include <math.h>

typedef __attribute__((ext_vector_type(16))) __bf16 v16bf;
typedef __attribute__((ext_vector_type(8)))  float  v8f;

#define WMMA_BF16(a, b, c) \
  __builtin_amdgcn_wmma_f32_16x16x32_bf16(false, (a), false, (b), (short)0, (c), false, false)

// Problem constants
constexpr int Tq  = 512;          // sequence length
constexpr int Bb  = 4;            // batch
constexpr int E   = 1024;         // embed dim
constexpr int Hh  = 16;           // heads
constexpr int HD  = 64;           // head dim
constexpr int Ng  = 2;            // ngrams
constexpr int F3  = 3 * E;        // 3072
constexpr int KH  = 512;          // buckets*heads
constexpr int TOKS = (1 + Ng) * Tq * Bb;   // 6144 tokens

// ---------------------------------------------------------------------------
// fp32 -> bf16 elementwise conversion
// ---------------------------------------------------------------------------
__global__ void cvt_f32_bf16(const float* __restrict__ in,
                             __bf16* __restrict__ out, int n) {
  int i = blockIdx.x * blockDim.x + threadIdx.x;
  if (i < n) out[i] = (__bf16)in[i];
}

__device__ inline v16bf ld16(const __bf16* p) {
  v16bf v;
#pragma unroll
  for (int j = 0; j < 16; ++j) v[j] = p[j];
  return v;
}

// ---------------------------------------------------------------------------
// C[M,N] = A[M,K] * B[N,K]^T + bias ; A f32 (cvt on the fly), B bf16.
// Block = 256 thr (8 waves): 64(M) x 128(N) tile; wave = 16(M) x 64(N).
// ---------------------------------------------------------------------------
__global__ __launch_bounds__(256) void gemm_nt_bf16(
    const float* __restrict__ A, const __bf16* __restrict__ Bw,
    const float* __restrict__ bias, void* __restrict__ Cout,
    int M, int N, int K, int out_bf16) {
  const int lane = threadIdx.x & 31;
  const int wave = threadIdx.x >> 5;
  const int lr   = lane & 15;
  const int hi   = (lane >> 4) & 1;
  const int m0   = blockIdx.y * 64  + (wave & 3) * 16;
  const int n0   = blockIdx.x * 128 + (wave >> 2) * 64;

  v8f acc[4] = {v8f{}, v8f{}, v8f{}, v8f{}};
  const float*  ap = A + (size_t)(m0 + lr) * K + hi * 16;
  const __bf16* bp[4];
#pragma unroll
  for (int t = 0; t < 4; ++t)
    bp[t] = Bw + (size_t)(n0 + t * 16 + lr) * K + hi * 16;

  for (int k0 = 0; k0 < K; k0 += 32) {
    v16bf a;
#pragma unroll
    for (int j = 0; j < 8; ++j) {
      a[j]     = (__bf16)ap[k0 + j];
      a[8 + j] = (__bf16)ap[k0 + 8 + j];
    }
#pragma unroll
    for (int t = 0; t < 4; ++t) {
      v16bf b = *(const v16bf*)(bp[t] + k0);
      acc[t] = WMMA_BF16(a, b, acc[t]);
    }
  }

#pragma unroll
  for (int r = 0; r < 8; ++r) {
    int row = m0 + hi * 8 + r;
#pragma unroll
    for (int t = 0; t < 4; ++t) {
      int col = n0 + t * 16 + lr;
      float v = acc[t][r] + bias[col];
      if (out_bf16) ((__bf16*)Cout)[(size_t)row * N + col] = (__bf16)v;
      else          ((float*) Cout)[(size_t)row * N + col] = v;
    }
  }
}

// ---------------------------------------------------------------------------
// Fused ngram attention (flash-style online softmax).
// grid = (T/64, B*H, 1+N). block = 128 thr (4 waves), wave = 16 query rows.
// qkv rows are tokens (t'*B + b), f in [0,3E): q|k|v each split to heads.
// ---------------------------------------------------------------------------
__global__ __launch_bounds__(128) void ngram_attn_kernel(
    const __bf16* __restrict__ qkv, const float* __restrict__ rel_all,
    const int* __restrict__ ib_main, const int* __restrict__ ib_rel,
    const float* __restrict__ mask_main, const float* __restrict__ mask_ng,
    float* __restrict__ attn_out) {
  const int lane = threadIdx.x & 31;
  const int wave = threadIdx.x >> 5;          // 0..3
  const int lr   = lane & 15;
  const int hi   = (lane >> 4) & 1;
  const int bh   = blockIdx.y;
  const int b    = bh >> 4;                   // / H
  const int h    = bh & 15;                   // % H
  const int z    = blockIdx.z;                // 0 = main, 1..N = ngram
  const int n    = z - 1;
  const int t0   = blockIdx.x * 64 + wave * 16;
  const int S_len = (z == 0) ? Tq : 2 * Tq;

  __shared__ alignas(32) __bf16 Ks[32][64];       // K tile  [s_local][c]
  __shared__ alignas(32) __bf16 Vt[64][32];       // V tile transposed [c][s_local]
  __shared__ alignas(32) __bf16 Plds[4][16][32];  // per-wave P strip

  // ---- Q fragments (A-layout 16x32, two K-windows of head dim 64) ----
  const int  tqr  = t0 + lr;
  const long qtok = (z == 0) ? (long)tqr : ((long)Tq + (long)n * Tq + tqr);
  const __bf16* qb = qkv + (size_t)(qtok * Bb + b) * F3 + h * HD;
  v16bf qa0, qa1;
#pragma unroll
  for (int j = 0; j < 8; ++j) {
    qa0[j]     = qb[hi * 16 + j];
    qa0[8 + j] = qb[hi * 16 + 8 + j];
    qa1[j]     = qb[32 + hi * 16 + j];
    qa1[8 + j] = qb[32 + hi * 16 + 8 + j];
  }

  float m_i[8], l_i[8];
#pragma unroll
  for (int r = 0; r < 8; ++r) { m_i[r] = -INFINITY; l_i[r] = 0.f; }
  v8f o[4] = {v8f{}, v8f{}, v8f{}, v8f{}};

  const int sl = threadIdx.x >> 2;            // 0..31 (staging row)
  const int c0 = (threadIdx.x & 3) * 16;      // staging col chunk

  for (int s0 = 0; s0 < S_len; s0 += 32) {
    // ---- cooperatively stage K (row-major) and V (transposed) in LDS ----
    {
      int  s    = s0 + sl;
      long stok = (z == 0 || s < Tq) ? (long)s
                                     : ((long)Tq + (long)n * Tq + (s - Tq));
      const __bf16* kp = qkv + (size_t)(stok * Bb + b) * F3 + E + h * HD + c0;
      v16bf kv = *(const v16bf*)kp;
      v16bf vv = *(const v16bf*)(kp + E);     // v lives E past k in the row
      *(v16bf*)&Ks[sl][c0] = kv;
#pragma unroll
      for (int j = 0; j < 16; ++j) Vt[c0 + j][sl] = vv[j];
    }
    __syncthreads();

    // ---- S = Q K^T (two 16-wide s-subtiles, two K-windows of 32) ----
    v8f sc0 = {}, sc1 = {};
    sc0 = WMMA_BF16(qa0, ld16(&Ks[lr][hi * 16]),           sc0);
    sc1 = WMMA_BF16(qa0, ld16(&Ks[16 + lr][hi * 16]),      sc1);
    sc0 = WMMA_BF16(qa1, ld16(&Ks[lr][32 + hi * 16]),      sc0);
    sc1 = WMMA_BF16(qa1, ld16(&Ks[16 + lr][32 + hi * 16]), sc1);

    // ---- scale + relative-bucket bias + mask, online softmax ----
    const int trow0 = t0 + hi * 8;
#pragma unroll
    for (int r = 0; r < 8; ++r) {
      const int t   = trow0 + r;
      const int s_a = s0 + lr;
      const int s_b = s0 + 16 + lr;
      long relrow;
      int  bk0, bk1;
      float mv0, mv1;
      if (z == 0) {
        relrow = (long)t * Bb + b;
        bk0 = ib_main[((size_t)b * Tq + t) * Tq + s_a];
        bk1 = ib_main[((size_t)b * Tq + t) * Tq + s_b];
        mv0 = mask_main[(size_t)t * Tq + s_a];
        mv1 = mask_main[(size_t)t * Tq + s_b];
      } else {
        relrow = ((long)Tq + (long)n * Tq + t) * Bb + b;
        bk0 = ib_rel[((size_t)b * Tq + t) * (2 * Tq) + s_a];
        bk1 = ib_rel[((size_t)b * Tq + t) * (2 * Tq) + s_b];
        mv0 = mask_ng[((size_t)n * Tq + t) * (2 * Tq) + s_a];
        mv1 = mask_ng[((size_t)n * Tq + t) * (2 * Tq) + s_b];
      }
      float x0 = sc0[r] * 0.125f + rel_all[(size_t)relrow * KH + bk0 * Hh + h] + mv0;
      float x1 = sc1[r] * 0.125f + rel_all[(size_t)relrow * KH + bk1 * Hh + h] + mv1;

      float mx = fmaxf(x0, x1);
#pragma unroll
      for (int off = 1; off < 16; off <<= 1)
        mx = fmaxf(mx, __shfl_xor(mx, off, 16));
      float mnew = fmaxf(m_i[r], mx);
      float p0 = __expf(x0 - mnew);
      float p1 = __expf(x1 - mnew);
      float ps = p0 + p1;
#pragma unroll
      for (int off = 1; off < 16; off <<= 1)
        ps += __shfl_xor(ps, off, 16);
      float cf = __expf(m_i[r] - mnew);
      l_i[r] = l_i[r] * cf + ps;
      m_i[r] = mnew;
#pragma unroll
      for (int f = 0; f < 4; ++f) o[f][r] *= cf;

      int prow = hi * 8 + r;
      Plds[wave][prow][lr]      = (__bf16)p0;
      Plds[wave][prow][lr + 16] = (__bf16)p1;
    }
    __syncthreads();

    // ---- O += P V  (P back from LDS in A-layout, V from transposed LDS) ----
    {
      v16bf pa;
      const __bf16* pp = &Plds[wave][lr][hi * 16];
#pragma unroll
      for (int j = 0; j < 8; ++j) { pa[j] = pp[j]; pa[8 + j] = pp[8 + j]; }
#pragma unroll
      for (int f = 0; f < 4; ++f)
        o[f] = WMMA_BF16(pa, ld16(&Vt[f * 16 + lr][hi * 16]), o[f]);
    }
    __syncthreads();
  }

  // ---- epilogue: normalize and store into attn scratch (token-major rows) --
#pragma unroll
  for (int r = 0; r < 8; ++r) {
    int  t   = t0 + hi * 8 + r;
    long tok = (z == 0) ? (long)t : ((long)Tq + (long)n * Tq + t);
    float inv = 1.0f / l_i[r];
    float* orow = attn_out + (size_t)(tok * Bb + b) * E + h * HD;
#pragma unroll
    for (int f = 0; f < 4; ++f) orow[f * 16 + lr] = o[f][r] * inv;
  }
}

// ---------------------------------------------------------------------------
// Host-side orchestration
// ---------------------------------------------------------------------------
extern "C" void kernel_launch(void* const* d_in, const int* in_sizes, int n_in,
                              void* d_out, int out_size, void* d_ws, size_t ws_size,
                              hipStream_t stream) {
  (void)in_sizes; (void)n_in; (void)out_size; (void)ws_size;
  const float* query     = (const float*)d_in[0];
  // d_in[1] = key, d_in[2] = value : unused by reference
  const float* mask_main = (const float*)d_in[3];
  const float* mask_ng   = (const float*)d_in[4];
  const int*   ib_main   = (const int*)  d_in[5];
  const int*   ib_rel    = (const int*)  d_in[6];
  const float* w_in      = (const float*)d_in[7];
  const float* b_in      = (const float*)d_in[8];
  const float* w_rel     = (const float*)d_in[9];
  const float* b_rel     = (const float*)d_in[10];
  const float* w_out     = (const float*)d_in[11];
  const float* b_out     = (const float*)d_in[12];

  char* ws = (char*)d_ws;
  __bf16* w_in_bf  = (__bf16*)(ws);                 //  6,291,456 B
  __bf16* w_rel_bf = (__bf16*)(ws +  6291456);      //  1,048,576 B
  __bf16* w_out_bf = (__bf16*)(ws +  7340032);      //  2,097,152 B
  __bf16* qkv_bf   = (__bf16*)(ws +  9437184);      // 37,748,736 B
  float*  rel_all  = (float*) (ws + 47185920);      // 12,582,912 B
  float*  attn_f   = (float*) (ws + 59768832);      // 25,165,824 B  (total ~85 MB)

  // weights -> bf16
  cvt_f32_bf16<<<(F3 * E + 255) / 256, 256, 0, stream>>>(w_in,  w_in_bf,  F3 * E);
  cvt_f32_bf16<<<(KH * E + 255) / 256, 256, 0, stream>>>(w_rel, w_rel_bf, KH * E);
  cvt_f32_bf16<<<(E * E + 255) / 256, 256, 0, stream>>>(w_out, w_out_bf, E * E);

  // qkv = query @ W_in^T + b_in  (bf16 out)
  gemm_nt_bf16<<<dim3(F3 / 128, TOKS / 64), 256, 0, stream>>>(
      query, w_in_bf, b_in, qkv_bf, TOKS, F3, E, 1);
  // rel_all = query @ W_rel^T + b_rel  (f32 out)
  gemm_nt_bf16<<<dim3(KH / 128, TOKS / 64), 256, 0, stream>>>(
      query, w_rel_bf, b_rel, rel_all, TOKS, KH, E, 0);

  // fused attention (main + 2 ngram streams)
  ngram_attn_kernel<<<dim3(Tq / 64, Bb * Hh, 1 + Ng), 128, 0, stream>>>(
      qkv_bf, rel_all, ib_main, ib_rel, mask_main, mask_ng, attn_f);

  // out = attn @ W_out^T + b_out  (f32 out, straight into d_out)
  gemm_nt_bf16<<<dim3(E / 128, TOKS / 64), 256, 0, stream>>>(
      attn_f, w_out_bf, b_out, (float*)d_out, TOKS, E, E, 0);
}
